// STEIConvNetMScaleCompactPROP_55662776156163
// MI455X (gfx1250) — matrix operand order repaired and checked by reading, not compiled, not run on hardware
//
#include <hip/hip_runtime.h>
#include <hip/hip_bf16.h>

// ---------------- problem constants (match reference) ----------------
#define N_NODES   20000
#define N_EDGES   320000
#define IN_DIM    16
#define HID       64
#define EDGE_DIM  16
#define EDGE_CAT  64          // EDGE_DIM * NUM_FILTERS
#define NUM_LAYERS 6
#define KIN_PAD   224         // 193 padded to 7*32
#define KOUT_PAD  128         // 112 padded to 4*32
#define WAVES     4           // waves per block (128 threads)

typedef __attribute__((ext_vector_type(16))) __bf16 v16bf;
typedef __attribute__((ext_vector_type(8)))  float  v8f;

static __device__ __forceinline__ unsigned short f2bf(float f) {
    unsigned int u = __float_as_uint(f);
    unsigned int r = u + 0x7FFFu + ((u >> 16) & 1u);   // round-to-nearest-even
    return (unsigned short)(r >> 16);
}
static __device__ __forceinline__ unsigned int pack2(unsigned short lo, unsigned short hi) {
    return (unsigned int)lo | ((unsigned int)hi << 16);
}

// A fragment (16x32 bf16): lane m=l&15 holds row m; VGPR v holds packed K pair.
static __device__ __forceinline__ v16bf loadAfragLDS(const unsigned short* buf, int rowstride,
                                                     int kbase, int lane) {
    int m = lane & 15, half = lane >> 4;
    union { unsigned int u[8]; v16bf v; } r;
#pragma unroll
    for (int v0 = 0; v0 < 8; ++v0) {
        int kp = (v0 < 4) ? (2 * v0 + 8 * half) : (16 + 2 * (v0 - 4) + 8 * half);
        r.u[v0] = *(const unsigned int*)&buf[m * rowstride + kbase + kp];
    }
    return r.v;
}

// B fragment (32x16 bf16) from row-major [K][64] bf16 weights in global (L2-resident).
static __device__ __forceinline__ v16bf loadBfragG(const unsigned short* W, int kbase,
                                                   int ntile, int lane) {
    int n = ntile * 16 + (lane & 15), half = lane >> 4;
    union { unsigned int u[8]; v16bf v; } r;
#pragma unroll
    for (int v0 = 0; v0 < 8; ++v0) {
        int kp = (v0 < 4) ? (2 * v0 + 8 * half) : (16 + 2 * (v0 - 4) + 8 * half);
        r.u[v0] = pack2(W[(kbase + kp) * 64 + n], W[(kbase + kp + 1) * 64 + n]);
    }
    return r.v;
}

static __device__ __forceinline__ v8f wmma_bf16(v16bf a, v16bf b, v8f c) {
    return __builtin_amdgcn_wmma_f32_16x16x32_bf16(false, a, false, b, (short)0, c, false, false);
}

// ---------------- weight prep: fp32 -> bf16, K-padded ----------------
__global__ void prep_weights_kernel(const float* W_e_in, const float* W_e_out, const float* W_n,
                                    unsigned short* Wein, unsigned short* Weout, unsigned short* Wn) {
    int i = blockIdx.x * blockDim.x + threadIdx.x;
    const int nein = 2 * KIN_PAD * 64, neout = KOUT_PAD * 64, nwn = 64 * 64;
    if (i < nein) {
        int s = i / (KIN_PAD * 64), rem = i % (KIN_PAD * 64);
        int kr = rem / 64, c = rem % 64;
        float v = (kr < 193) ? W_e_in[(s * 193 + kr) * 64 + c] : 0.0f;
        Wein[i] = f2bf(v);
    } else if (i < nein + neout) {
        int j = i - nein; int kr = j / 64, c = j % 64;
        float v = (kr < 112) ? W_e_out[kr * 64 + c] : 0.0f;
        Weout[j] = f2bf(v);
    } else if (i < nein + neout + nwn) {
        int j = i - nein - neout;
        Wn[j] = f2bf(W_n[j]);
    }
}

// ---------------- node embedding: h0 = inputs @ W_emb ----------------
__global__ void embed_kernel(const float* inputs, const float* W_emb, float* h) {
    int i = blockIdx.x * blockDim.x + threadIdx.x;   // over N_NODES*HID
    if (i >= N_NODES * HID) return;
    int n = i >> 6, c = i & 63;
    float acc = 0.0f;
#pragma unroll
    for (int k = 0; k < IN_DIM; ++k) acc += inputs[n * IN_DIM + k] * W_emb[k * HID + c];
    h[i] = acc;
}

// ---------------- fused edge pipeline (2 IN layers + OUT layer + scatter) ----------------
__global__ void __launch_bounds__(WAVES * 32)
edge_kernel(const float* __restrict__ h, const float* __restrict__ e_feats,
            const float* __restrict__ rain0, const int* __restrict__ src,
            const int* __restrict__ dst, const unsigned short* __restrict__ Wein,
            const unsigned short* __restrict__ Weout, float* __restrict__ agg, int layer) {
    __shared__ unsigned short ein[WAVES][16 * KIN_PAD];    // 7168 B / wave
    __shared__ unsigned short eout[WAVES][16 * KOUT_PAD];  // 4096 B / wave
    __shared__ unsigned short etile[WAVES][16 * 16];       // 512 B / wave

    const int lane = threadIdx.x & 31;
    const int wave = threadIdx.x >> 5;
    const int half = lane >> 4;
    const int numTiles = (N_EDGES + 15) / 16;

    for (int base = blockIdx.x * WAVES; base < numTiles; base += gridDim.x * WAVES) {
        const int tile = base + wave;
        const bool active = tile < numTiles;
        const int e0 = tile * 16;

        // ---- stage the 16x224 bf16 edge_in block in LDS ----
        if (active) {
            // h[src] rows -> cols [0,64), h[dst] rows -> cols [64,128)
#pragma unroll 4
            for (int p = 0; p < 16; ++p) {
                int m = p;
                int e = e0 + m;
                int node = half ? dst[e] : src[e];
                int c4 = (lane & 15) * 4;
                const float4 hv = *(const float4*)&h[node * HID + c4];
                unsigned int w0 = pack2(f2bf(hv.x), f2bf(hv.y));
                unsigned int w1 = pack2(f2bf(hv.z), f2bf(hv.w));
                unsigned int* dp = (unsigned int*)&ein[wave][m * KIN_PAD + half * 64 + c4];
                dp[0] = w0; dp[1] = w1;
            }
            // e_t tile -> etile and replicated x4 into cols [128,192)
#pragma unroll
            for (int p = 0; p < 8; ++p) {
                int idx = p * 32 + lane;                 // 0..255 = 16 edges x 16 dims
                int m = idx >> 4, d = idx & 15;
                float v = e_feats[((e0 + m) * EDGE_DIM + d) * NUM_LAYERS + layer];
                unsigned short b = f2bf(v);
                etile[wave][m * 16 + d] = b;
                ein[wave][m * KIN_PAD + 128 + d] = b;
                ein[wave][m * KIN_PAD + 144 + d] = b;
                ein[wave][m * KIN_PAD + 160 + d] = b;
                ein[wave][m * KIN_PAD + 176 + d] = b;
            }
            // rain[src] -> col 192, zeros -> cols [193,224)
#pragma unroll
            for (int p = 0; p < 16; ++p) {
                int idx = p * 32 + lane;                 // 0..511 = 16 rows x 32 cols
                int m = idx >> 5, c = idx & 31;
                unsigned short b = 0;
                if (c == 0) b = f2bf(rain0[src[e0 + m] * NUM_LAYERS + layer]);
                ein[wave][m * KIN_PAD + 192 + c] = b;
            }
        }
        __syncthreads();

        v8f C[4];
        const v8f vzero = {0, 0, 0, 0, 0, 0, 0, 0};

        // ---- two IN layers: ef = relu([h_s|h_d|ef|rain] @ W_e_in[s]) ----
        for (int s = 0; s < 2; ++s) {
            C[0] = vzero; C[1] = vzero; C[2] = vzero; C[3] = vzero;
            const unsigned short* W = Wein + s * (KIN_PAD * 64);
            for (int kc = 0; kc < 7; ++kc) {
                v16bf A = loadAfragLDS(&ein[wave][0], KIN_PAD, kc * 32, lane);
#pragma unroll
                for (int nt = 0; nt < 4; ++nt) {
                    v16bf B = loadBfragG(W, kc * 32, nt, lane);
                    C[nt] = wmma_bf16(A, B, C[nt]);
                }
            }
            __syncthreads();   // all reads of ein middle block done before overwrite
            if (active) {
#pragma unroll
                for (int nt = 0; nt < 4; ++nt)
#pragma unroll
                    for (int r = 0; r < 8; ++r) {
                        int m = r + 8 * half;
                        int col = nt * 16 + (lane & 15);
                        float v = C[nt][r]; v = v > 0.0f ? v : 0.0f;
                        if (s == 0) ein[wave][m * KIN_PAD + 128 + col] = f2bf(v);
                        else        eout[wave][m * KOUT_PAD + col]     = f2bf(v);
                    }
            }
            __syncthreads();
        }

        // ---- finish OUT-layer input: cols [64,112) = tile(e_t,3), [112,128) = 0 ----
        if (active) {
#pragma unroll 4
            for (int p = 0; p < 32; ++p) {
                int idx = p * 32 + lane;                 // 0..1023 = 16 rows x 64 cols
                int m = idx >> 6, c = idx & 63;
                unsigned short b = (c < 48) ? etile[wave][m * 16 + (c & 15)] : (unsigned short)0;
                eout[wave][m * KOUT_PAD + 64 + c] = b;
            }
        }
        __syncthreads();

        // ---- OUT layer: m = relu(edge_in @ W_e_out), scatter-add into agg[dst] ----
        C[0] = vzero; C[1] = vzero; C[2] = vzero; C[3] = vzero;
        for (int kc = 0; kc < 4; ++kc) {
            v16bf A = loadAfragLDS(&eout[wave][0], KOUT_PAD, kc * 32, lane);
#pragma unroll
            for (int nt = 0; nt < 4; ++nt) {
                v16bf B = loadBfragG(Weout, kc * 32, nt, lane);
                C[nt] = wmma_bf16(A, B, C[nt]);
            }
        }
        if (active) {
#pragma unroll
            for (int nt = 0; nt < 4; ++nt)
#pragma unroll
                for (int r = 0; r < 8; ++r) {
                    int m = r + 8 * half;
                    float v = C[nt][r]; v = v > 0.0f ? v : 0.0f;
                    int node = dst[e0 + m];
                    atomicAdd(&agg[node * HID + nt * 16 + (lane & 15)], v);
                }
        }
        __syncthreads();
    }
}

// ---------------- node update: h = tanh(agg @ W_n + rain*w_r); rain_out = h @ w_out ----
__global__ void __launch_bounds__(WAVES * 32)
node_kernel(const float* __restrict__ agg, const float* __restrict__ rain0,
            const unsigned short* __restrict__ Wn, const float* __restrict__ w_r,
            const float* __restrict__ w_out, float* __restrict__ h,
            float* __restrict__ out, int layer) {
    __shared__ float hstage[WAVES][16 * HID];            // 4 KB / wave

    const int lane = threadIdx.x & 31;
    const int wave = threadIdx.x >> 5;
    const int half = lane >> 4;
    const int numTiles = (N_NODES + 15) / 16;

    // hoist B fragments of W_n (2 K-chunks x 4 N-tiles)
    v16bf Bn[2][4];
#pragma unroll
    for (int kc = 0; kc < 2; ++kc)
#pragma unroll
        for (int nt = 0; nt < 4; ++nt) Bn[kc][nt] = loadBfragG(Wn, kc * 32, nt, lane);

    for (int base = blockIdx.x * WAVES; base < numTiles; base += gridDim.x * WAVES) {
        const int tile = base + wave;
        const bool active = tile < numTiles;
        const int n0 = tile * 16;

        const v8f vzero = {0, 0, 0, 0, 0, 0, 0, 0};
        v8f C[4] = {vzero, vzero, vzero, vzero};

        for (int kc = 0; kc < 2; ++kc) {
            union { unsigned int u[8]; v16bf v; } a;
            int m = lane & 15;
            int node = n0 + m;
            const float* ap = active ? &agg[node * HID] : &agg[0];
#pragma unroll
            for (int v0 = 0; v0 < 8; ++v0) {
                int kp = (v0 < 4) ? (2 * v0 + 8 * half) : (16 + 2 * (v0 - 4) + 8 * half);
                int K = kc * 32 + kp;
                a.u[v0] = pack2(f2bf(ap[K]), f2bf(ap[K + 1]));
            }
#pragma unroll
            for (int nt = 0; nt < 4; ++nt) C[nt] = wmma_bf16(a.v, Bn[kc][nt], C[nt]);
        }

        if (active) {
#pragma unroll
            for (int nt = 0; nt < 4; ++nt)
#pragma unroll
                for (int r = 0; r < 8; ++r) {
                    int m = r + 8 * half;
                    int node = n0 + m;
                    int col = nt * 16 + (lane & 15);
                    float v = C[nt][r] + rain0[node * NUM_LAYERS + layer] * w_r[col];
                    v = tanhf(v);
                    h[node * HID + col] = v;
                    hstage[wave][m * HID + col] = v;
                }
        }
        __syncthreads();
        if (active && lane < 16) {
            int node = n0 + lane;
            float s = 0.0f;
#pragma unroll 8
            for (int k = 0; k < HID; ++k) s += hstage[wave][lane * HID + k] * w_out[k];
            out[node * NUM_LAYERS + layer] = s;   // rains.T layout [N, L]
        }
        __syncthreads();
    }
}

// ---------------- launcher ----------------
extern "C" void kernel_launch(void* const* d_in, const int* in_sizes, int n_in,
                              void* d_out, int out_size, void* d_ws, size_t ws_size,
                              hipStream_t stream) {
    const float* inputs  = (const float*)d_in[0];
    const float* e_feats = (const float*)d_in[1];
    const float* rain0   = (const float*)d_in[2];
    const int*   src     = (const int*)d_in[3];
    const int*   dst     = (const int*)d_in[4];
    const float* W_emb   = (const float*)d_in[5];
    const float* W_e_in  = (const float*)d_in[6];
    const float* W_e_out = (const float*)d_in[7];
    const float* W_n     = (const float*)d_in[8];
    const float* w_r     = (const float*)d_in[9];
    const float* w_out   = (const float*)d_in[10];
    float* out = (float*)d_out;

    // workspace carve-up (256B aligned)
    char* ws = (char*)d_ws;
    size_t off = 0;
    auto alloc = [&](size_t bytes) { char* p = ws + off; off = (off + bytes + 255) & ~(size_t)255; return p; };
    float*          h    = (float*)alloc((size_t)N_NODES * HID * sizeof(float));
    float*          agg  = (float*)alloc((size_t)N_NODES * HID * sizeof(float));
    unsigned short* Wein = (unsigned short*)alloc((size_t)2 * KIN_PAD * 64 * sizeof(unsigned short));
    unsigned short* Weout= (unsigned short*)alloc((size_t)KOUT_PAD * 64 * sizeof(unsigned short));
    unsigned short* Wn   = (unsigned short*)alloc((size_t)64 * 64 * sizeof(unsigned short));
    (void)ws_size; (void)in_sizes; (void)n_in; (void)out_size;

    const int prepN = 2 * KIN_PAD * 64 + KOUT_PAD * 64 + 64 * 64;
    prep_weights_kernel<<<(prepN + 255) / 256, 256, 0, stream>>>(W_e_in, W_e_out, W_n, Wein, Weout, Wn);
    embed_kernel<<<(N_NODES * HID + 255) / 256, 256, 0, stream>>>(inputs, W_emb, h);

    const int edgeTiles = (N_EDGES + 15) / 16;
    const int nodeTiles = (N_NODES + 15) / 16;
    const int edgeBlocks = (edgeTiles + WAVES - 1) / WAVES;
    const int nodeBlocks = (nodeTiles + WAVES - 1) / WAVES;

    for (int t = 0; t < NUM_LAYERS; ++t) {
        hipMemsetAsync(agg, 0, (size_t)N_NODES * HID * sizeof(float), stream);
        edge_kernel<<<edgeBlocks, WAVES * 32, 0, stream>>>(h, e_feats, rain0, src, dst,
                                                           Wein, Weout, agg, t);
        node_kernel<<<nodeBlocks, WAVES * 32, 0, stream>>>(agg, rain0, Wn, w_r, w_out,
                                                           h, out, t);
    }
}
